// BisineNetwork_19292993094058
// MI455X (gfx1250) — matrix operand order repaired and checked by reading, not compiled, not run on hardware
//
#include <hip/hip_runtime.h>

typedef _Float16 v16h __attribute__((ext_vector_type(16)));
typedef _Float16 v8h  __attribute__((ext_vector_type(8)));
typedef float    v8f  __attribute__((ext_vector_type(8)));
typedef float    v4f  __attribute__((ext_vector_type(4)));
typedef int      v4i  __attribute__((ext_vector_type(4)));

#define DD     256      // D
#define CC     1000     // classes
#define KKU    4        // K units per class
#define PU     515      // 2*D+3
#define NROWS  16384
#define CKCOLS 4000     // C*K columns

// ---------------- workspace layout (bytes) ----------------
#define WS_XH   0ul                                  // 16384*256 f16 = 8,388,608
#define WS_W1H  (WS_XH  + (size_t)NROWS*DD*2)        // 4000*256 f16  = 2,048,000
#define WS_W2H  (WS_W1H + (size_t)CKCOLS*DD*2)       // contiguous after w1h
#define WS_A    (WS_W2H + (size_t)CKCOLS*DD*2)       // 4000 f32
#define WS_B1   (WS_A   + (size_t)CKCOLS*4)
#define WS_B2   (WS_B1  + (size_t)CKCOLS*4)

// CDNA5 async global->LDS path (ASYNCcnt), with compile-safe fallback.
#if __has_builtin(__builtin_amdgcn_global_load_async_to_lds_b128) && \
    __has_builtin(__builtin_amdgcn_s_wait_asynccnt)
#define USE_ASYNC_LDS 1
#else
#define USE_ASYNC_LDS 0
#endif

typedef __attribute__((address_space(1))) v4i g_v4i;
typedef __attribute__((address_space(3))) v4i l_v4i;

__device__ __forceinline__ void cp16_to_lds(const _Float16* g, _Float16* l) {
#if USE_ASYNC_LDS
  __builtin_amdgcn_global_load_async_to_lds_b128(
      (g_v4i*)g, (l_v4i*)l, 0, 0);
#else
  *(v8h*)l = *(const v8h*)g;
#endif
}

__device__ __forceinline__ void stage_fence() {
#if USE_ASYNC_LDS
  __builtin_amdgcn_s_wait_asynccnt(0);
#endif
}

// ---------------- prep: x (f32) -> xh (f16) ----------------
__global__ __launch_bounds__(256) void cvt_x_kernel(const float* __restrict__ x,
                                                    _Float16* __restrict__ xh) {
  int i = (blockIdx.x * 256 + threadIdx.x) * 8;
  v4f a = *(const v4f*)(x + i);
  v4f b = *(const v4f*)(x + i + 4);
  v8h h;
  h[0] = (_Float16)a[0]; h[1] = (_Float16)a[1];
  h[2] = (_Float16)a[2]; h[3] = (_Float16)a[3];
  h[4] = (_Float16)b[0]; h[5] = (_Float16)b[1];
  h[6] = (_Float16)b[2]; h[7] = (_Float16)b[3];
  *(v8h*)(xh + i) = h;
}

// ---------------- prep: params -> w1h/w2h (f16, K-contiguous) + a/b1/b2 ----------------
__global__ __launch_bounds__(256) void prep_params_kernel(const float* __restrict__ params,
                                                          _Float16* __restrict__ w1h,
                                                          _Float16* __restrict__ w2h,
                                                          float* __restrict__ av,
                                                          float* __restrict__ b1v,
                                                          float* __restrict__ b2v) {
  int j = blockIdx.x;            // column index: j = c*4 + k, 0..3999
  int c = j >> 2;
  int k = j & 3;
  int d = threadIdx.x;           // 0..255
  const float* base = params + (size_t)c * (KKU * PU) + (size_t)k * PU;
  // p[c][k]: [0]=a, [1..256]=w1, [257]=b1, [258..513]=w2, [514]=b2
  w1h[(size_t)j * DD + d] = (_Float16)base[1 + d];
  w2h[(size_t)j * DD + d] = (_Float16)base[2 + DD + d];
  if (d == 0) {
    av[j]  = base[0];
    b1v[j] = base[1 + DD];
    b2v[j] = base[2 + 2 * DD];
  }
}

// ---------------- main fused GEMM + bisine kernel ----------------
// Block: 256 threads = 8 waves. Wave w handles rows [blockIdx.x*128 + w*16, +16),
// columns [blockIdx.y*80, +80) (5 WMMA N-subtiles) for both w1 and w2 GEMMs.
// K is processed in 4 stages of 64 (two 32-wide WMMA steps per stage),
// double-buffered through LDS with async global->LDS staging.
#define BSTRIDE 72   // 64 K-halves + 8 pad halves -> conflict-free b128 DS reads

__global__ __launch_bounds__(256) void bisine_main_kernel(
    const _Float16* __restrict__ xh,
    const _Float16* __restrict__ w1h,   // w2h = w1h + CKCOLS*DD (contiguous)
    const float* __restrict__ av,
    const float* __restrict__ b1v,
    const float* __restrict__ b2v,
    float* __restrict__ out) {
  __shared__ _Float16 ldsB[2][2 * 80 * BSTRIDE];   // 2 x 23,040 B

  const int tid   = threadIdx.x;
  const int wave  = tid >> 5;
  const int lane  = tid & 31;
  const int col16 = lane & 15;
  const int hi    = lane >> 4;                     // 0: lanes 0-15, 1: lanes 16-31

  const int m_base = (blockIdx.x * 8 + wave) * 16; // row tile base
  const int n_base = blockIdx.y * 80;              // column group base

  v8f acc1[5] = {};
  v8f acc2[5] = {};

  // A-operand per-lane source: row M = lane&15, K chunks at hi*8 and hi*8+16
  const _Float16* xrow = xh + (size_t)(m_base + col16) * DD;
  const int a_koff = hi * 8;
  const int b_koff = hi * 8;

  // Branch-free cooperative stage of one K-pair (64 halves) of both weight
  // matrices for 80 columns: 1280 16B chunks = exactly 5 per thread.
  auto stage = [&](int ktp, _Float16* buf) {
#pragma unroll
    for (int q = 0; q < 5; ++q) {
      const int idx = q * 256 + tid;         // 0..1279
      const int mat = idx >= 640;            // 0: w1, 1: w2
      const int rem = idx - mat * 640;
      const int col = rem >> 3;              // 0..79
      const int seg = rem & 7;               // 8 x 8-half segments
      const _Float16* src = w1h + (size_t)mat * ((size_t)CKCOLS * DD) +
                            (size_t)(n_base + col) * DD + ktp * 64 + seg * 8;
      _Float16* dst = buf + mat * (80 * BSTRIDE) + col * BSTRIDE + seg * 8;
      cp16_to_lds(src, dst);
    }
  };

  stage(0, ldsB[0]);
  stage_fence();
  __syncthreads();

  for (int ktp = 0; ktp < 4; ++ktp) {
    if (ktp < 3) stage(ktp + 1, ldsB[(ktp + 1) & 1]);  // overlaps with compute

    const _Float16* buf = ldsB[ktp & 1];
#pragma unroll
    for (int sub = 0; sub < 2; ++sub) {
      const int kbase = ktp * 64 + sub * 32;

      union { v16h v; v8h h[2]; } A;
      A.h[0] = *(const v8h*)(xrow + kbase + a_koff);
      A.h[1] = *(const v8h*)(xrow + kbase + a_koff + 16);

#pragma unroll
      for (int s = 0; s < 5; ++s) {
        const int bidx = (s * 16 + col16) * BSTRIDE + sub * 32 + b_koff;
        union { v16h v; v8h h[2]; } B1, B2;
        B1.h[0] = *(const v8h*)(&buf[bidx]);
        B1.h[1] = *(const v8h*)(&buf[bidx + 16]);
        B2.h[0] = *(const v8h*)(&buf[80 * BSTRIDE + bidx]);
        B2.h[1] = *(const v8h*)(&buf[80 * BSTRIDE + bidx + 16]);
        acc1[s] = __builtin_amdgcn_wmma_f32_16x16x32_f16(
            false, A.v, false, B1.v, (short)0, acc1[s], false, false);
        acc2[s] = __builtin_amdgcn_wmma_f32_16x16x32_f16(
            false, A.v, false, B2.v, (short)0, acc2[s], false, false);
      }
    }
    stage_fence();      // our async writes to the other buffer are done
    __syncthreads();    // everyone's writes visible; reads of current buf done
  }

  // Epilogue: u -> a*sin(u1)*sin(u2), reduce over k (4 consecutive lanes), store.
  // D-matrix layout: VGPR r holds row M = r + hi*8, column N = lane&15.
#pragma unroll
  for (int s = 0; s < 5; ++s) {
    const int j    = n_base + s * 16 + col16;   // global column = c*4 + k
    const float ac = av[j];
    const float o1 = b1v[j];
    const float o2 = b2v[j];
    const int cidx = j >> 2;
#pragma unroll
    for (int r = 0; r < 8; ++r) {
      float u1 = acc1[s][r] + o1;
      float u2 = acc2[s][r] + o2;
      float v  = ac * __sinf(u1) * __sinf(u2);
      v += __shfl_xor(v, 1, 32);
      v += __shfl_xor(v, 2, 32);
      if ((lane & 3) == 0) {
        const int row = m_base + r + hi * 8;
        out[(size_t)row * CC + cidx] = v;
      }
    }
  }
}

extern "C" void kernel_launch(void* const* d_in, const int* in_sizes, int n_in,
                              void* d_out, int out_size, void* d_ws, size_t ws_size,
                              hipStream_t stream) {
  const float* x      = (const float*)d_in[0];   // 16384*256 f32
  const float* params = (const float*)d_in[1];   // 1000*2060 f32
  float* out          = (float*)d_out;           // 16384*1000 f32

  char* ws = (char*)d_ws;
  _Float16* xh  = (_Float16*)(ws + WS_XH);
  _Float16* w1h = (_Float16*)(ws + WS_W1H);
  _Float16* w2h = (_Float16*)(ws + WS_W2H);
  float*    avp = (float*)(ws + WS_A);
  float*    b1p = (float*)(ws + WS_B1);
  float*    b2p = (float*)(ws + WS_B2);

  // 1) x -> f16   (16384*256 / 8 elems per thread / 256 per block = 2048 blocks)
  cvt_x_kernel<<<dim3((NROWS * DD) / (8 * 256)), dim3(256), 0, stream>>>(x, xh);

  // 2) params -> f16 weight matrices + f32 scalars (one block per column)
  prep_params_kernel<<<dim3(CKCOLS), dim3(256), 0, stream>>>(params, w1h, w2h,
                                                             avp, b1p, b2p);

  // 3) fused WMMA GEMMs + bisine epilogue
  //    grid: 16384/(8*16) = 128 row-blocks, 4000/80 = 50 column groups
  bisine_main_kernel<<<dim3(128, 50), dim3(256), 0, stream>>>(
      xh, w1h, avp, b1p, b2p, out);
}
